// MEGNet_2697239462209
// MI455X (gfx1250) — compile-verified, matrix operand order checked by reading, not run on hardware
//
#include <hip/hip_runtime.h>
#include <hip/hip_bf16.h>

typedef __attribute__((ext_vector_type(16))) _Float16 v16h;
typedef __attribute__((ext_vector_type(8)))  float    v8f;

union AFrag { v16h v; _Float16 h[16]; };
union CFrag { v8f  v; float    f[8];  };

// CDNA5 16-bit WMMA operand layout: element e (0..15) of the v16h fragment for
// this lane maps to K index:
//   lanes 0-15 : K = {0..7, 16..23};  lanes 16-31 : K = {8..15, 24..31}
__device__ __forceinline__ int kmap(int e, int hf) {
  return ((e & 8) ? 16 : 0) + (hf << 3) + (((e >> 1) & 3) << 1) + (e & 1);
}

struct MlpW {
  const float *W0, *b0, *W1, *b1, *W2, *b2;
  float *Weff, *beff;
  int I;
};

// ---------------------------------------------------------------- utility ----
__global__ void k_zero(float* p, long n) {
  long i  = (long)blockIdx.x * blockDim.x + threadIdx.x;
  long st = (long)gridDim.x * blockDim.x;
  for (; i < n; i += st) p[i] = 0.f;
}

// Collapse 3 stacked linear layers (no activation in reference) into one:
// Weff = W2*W1*W0 (32 x I), beff = W2*(W1*b0 + b1) + b2.
__global__ void k_setup(MlpW a, MlpW b, MlpW c) {
  __shared__ float T[32 * 128];
  __shared__ float tb[32];
  MlpW ms[3] = {a, b, c};
  for (int q = 0; q < 3; ++q) {
    MlpW m = ms[q];
    int I = m.I;
    __syncthreads();  // protect T/tb reuse across iterations
    for (int idx = threadIdx.x; idx < 32 * I; idx += blockDim.x) {
      int i = idx / I, j = idx % I;
      float s = 0.f;
      for (int k = 0; k < 32; ++k) s += m.W1[i * 32 + k] * m.W0[k * I + j];
      T[idx] = s;
    }
    if (threadIdx.x < 32) {
      float s = m.b1[threadIdx.x];
      for (int k = 0; k < 32; ++k) s += m.W1[threadIdx.x * 32 + k] * m.b0[k];
      tb[threadIdx.x] = s;
    }
    __syncthreads();
    for (int idx = threadIdx.x; idx < 32 * I; idx += blockDim.x) {
      int i = idx / I, j = idx % I;
      float s = 0.f;
      for (int k = 0; k < 32; ++k) s += m.W2[i * 32 + k] * T[k * I + j];
      m.Weff[idx] = s;
    }
    if (threadIdx.x < 32) {
      float s = m.b2[threadIdx.x];
      for (int k = 0; k < 32; ++k) s += m.W2[threadIdx.x * 32 + k] * tb[k];
      m.beff[threadIdx.x] = s;
    }
  }
}

// ------------------------------------------------------------- edge block ----
// Per wave: 16 edges, both directions fused. x_fwd=[nf[s],nf[d],ef,g] (128),
// x_bwd=[nf[d],nf[s],ef,g]. y = x @ Weff^T + beff via v_wmma_f32_16x16x32_f16
// (4 K-chunks x 2 N-tiles x 2 directions = 16 WMMA per tile).
// Writes ef_out = 0.5*(yf+yb)+ef, scatters yf->acc[dst], yb->acc[src],
// counts, and block-reduced sum of all 2E ef_new rows into e_sum.
__global__ void __launch_bounds__(256) k_edge(
    const float* __restrict__ nf, const int* __restrict__ src,
    const int* __restrict__ dst, const float* __restrict__ ef,
    const float* __restrict__ gf, const float* __restrict__ Weff,
    const float* __restrict__ beff, float* __restrict__ acc,
    float* __restrict__ cnt, float* __restrict__ ef_out,
    float* __restrict__ e_sum, int E)
{
  __shared__ _Float16 sA[8][16][96];   // per-wave: [row][src nf | dst nf | ef]
  __shared__ _Float16 sW[32][128];     // collapsed edge weight (row = out feat)
  __shared__ float    sB[32];
  __shared__ _Float16 sG[32];
  __shared__ float    sEsum[32];
  __shared__ int      sSrc[8][16], sDst[8][16];

  const int tid = threadIdx.x, wave = tid >> 5, lane = tid & 31;
  for (int i = tid; i < 32 * 128; i += 256) sW[i >> 7][i & 127] = (_Float16)Weff[i];
  if (tid < 32) { sB[tid] = beff[tid]; sG[tid] = (_Float16)gf[tid]; sEsum[tid] = 0.f; }

  const int ebase = blockIdx.x * 128 + wave * 16;
  if (lane < 16) {
    int e  = ebase + lane;
    int ec = e < E ? e : (E - 1);
    int s = src[ec], d = dst[ec];
    sSrc[wave][lane] = s;
    sDst[wave][lane] = d;
    if (e < E) { atomicAdd(&cnt[d], 1.f); atomicAdd(&cnt[s], 1.f); }
  }
  __syncthreads();

  for (int t = lane; t < 512; t += 32) {
    int r = t >> 5, f = t & 31;
    int e = ebase + r;
    int ec = e < E ? e : (E - 1);
    int s = sSrc[wave][r], d = sDst[wave][r];
    sA[wave][r][f]      = (_Float16)nf[s * 32 + f];
    sA[wave][r][32 + f] = (_Float16)nf[d * 32 + f];
    sA[wave][r][64 + f] = (_Float16)ef[ec * 32 + f];
  }
  __syncthreads();

  const int hf = lane >> 4, l16 = lane & 15;
  CFrag cf[2], cb[2];
  for (int t = 0; t < 2; ++t)
    for (int i = 0; i < 8; ++i) { cf[t].f[i] = 0.f; cb[t].f[i] = 0.f; }

  for (int c = 0; c < 4; ++c) {
    AFrag af, ab, b0, b1;
    for (int e2 = 0; e2 < 16; ++e2) {
      int k = kmap(e2, hf);
      if (c == 3) {
        _Float16 g = sG[k];
        af.h[e2] = g; ab.h[e2] = g;
      } else {
        int cbk = (c == 0) ? 1 : ((c == 1) ? 0 : 2);  // swap src/dst segments
        af.h[e2] = sA[wave][l16][c * 32 + k];
        ab.h[e2] = sA[wave][l16][cbk * 32 + k];
      }
      b0.h[e2] = sW[l16][c * 32 + k];
      b1.h[e2] = sW[16 + l16][c * 32 + k];
    }
    cf[0].v = __builtin_amdgcn_wmma_f32_16x16x32_f16(false, af.v, false, b0.v, (short)0, cf[0].v, false, false);
    cf[1].v = __builtin_amdgcn_wmma_f32_16x16x32_f16(false, af.v, false, b1.v, (short)0, cf[1].v, false, false);
    cb[0].v = __builtin_amdgcn_wmma_f32_16x16x32_f16(false, ab.v, false, b0.v, (short)0, cb[0].v, false, false);
    cb[1].v = __builtin_amdgcn_wmma_f32_16x16x32_f16(false, ab.v, false, b1.v, (short)0, cb[1].v, false, false);
  }

  for (int nt = 0; nt < 2; ++nt) {
    int n = nt * 16 + l16;
    float bia = sB[n];
    float lsum = 0.f;
    for (int r = 0; r < 8; ++r) {
      int m = r + 8 * hf;
      int e = ebase + m;
      float yf = cf[nt].f[r] + bia;
      float yb = cb[nt].f[r] + bia;
      if (e < E) {
        lsum += yf + yb;
        int s = sSrc[wave][m], d = sDst[wave][m];
        atomicAdd(&acc[(size_t)d * 32 + n], yf);
        atomicAdd(&acc[(size_t)s * 32 + n], yb);
        ef_out[(size_t)e * 32 + n] = 0.5f * (yf + yb) + ef[(size_t)e * 32 + n];
      }
    }
    atomicAdd(&sEsum[n], lsum);
  }
  __syncthreads();
  if (tid < 32) atomicAdd(&e_sum[tid], sEsum[tid]);
}

// ------------------------------------------------------------- node block ----
// x = [nf | segment_mean(acc,cnt) | g] (96) -> y = x@Weff^T+beff,
// nf_out = y + nf, block-reduced sum of y into n_sum.
__global__ void __launch_bounds__(256) k_node(
    const float* __restrict__ nf, const float* __restrict__ acc,
    const float* __restrict__ cnt, const float* __restrict__ gf,
    const float* __restrict__ Weff, const float* __restrict__ beff,
    float* __restrict__ nf_out, float* __restrict__ n_sum, int N)
{
  __shared__ _Float16 sA[8][16][64];
  __shared__ _Float16 sW[32][96];
  __shared__ float    sB[32];
  __shared__ _Float16 sG[32];
  __shared__ float    sNsum[32];

  const int tid = threadIdx.x, wave = tid >> 5, lane = tid & 31;
  for (int i = tid; i < 32 * 96; i += 256) sW[i / 96][i % 96] = (_Float16)Weff[i];
  if (tid < 32) { sB[tid] = beff[tid]; sG[tid] = (_Float16)gf[tid]; sNsum[tid] = 0.f; }

  const int nbase = blockIdx.x * 128 + wave * 16;
  for (int t = lane; t < 512; t += 32) {
    int r = t >> 5, f = t & 31;
    int row = nbase + r;
    int rc  = row < N ? row : (N - 1);
    float cc = cnt[rc];
    cc = cc < 1.f ? 1.f : cc;
    sA[wave][r][f]      = (_Float16)nf[(size_t)rc * 32 + f];
    sA[wave][r][32 + f] = (_Float16)(acc[(size_t)rc * 32 + f] / cc);
  }
  __syncthreads();

  const int hf = lane >> 4, l16 = lane & 15;
  CFrag cc2[2];
  for (int t = 0; t < 2; ++t)
    for (int i = 0; i < 8; ++i) cc2[t].f[i] = 0.f;

  for (int c = 0; c < 3; ++c) {
    AFrag af, b0, b1;
    for (int e2 = 0; e2 < 16; ++e2) {
      int k = kmap(e2, hf);
      af.h[e2] = (c == 2) ? sG[k] : sA[wave][l16][c * 32 + k];
      b0.h[e2] = sW[l16][c * 32 + k];
      b1.h[e2] = sW[16 + l16][c * 32 + k];
    }
    cc2[0].v = __builtin_amdgcn_wmma_f32_16x16x32_f16(false, af.v, false, b0.v, (short)0, cc2[0].v, false, false);
    cc2[1].v = __builtin_amdgcn_wmma_f32_16x16x32_f16(false, af.v, false, b1.v, (short)0, cc2[1].v, false, false);
  }

  for (int nt = 0; nt < 2; ++nt) {
    int n = nt * 16 + l16;
    float bia = sB[n];
    float lsum = 0.f;
    for (int r = 0; r < 8; ++r) {
      int m = r + 8 * hf;
      int row = nbase + m;
      float y = cc2[nt].f[r] + bia;
      if (row < N) {
        lsum += y;
        nf_out[(size_t)row * 32 + n] = y + nf[(size_t)row * 32 + n];
      }
    }
    atomicAdd(&sNsum[n], lsum);
  }
  __syncthreads();
  if (tid < 32) atomicAdd(&n_sum[tid], sNsum[tid]);
}

// ----------------------------------------------------------- global block ----
__global__ void k_global(const float* gf, const float* Weff, const float* beff,
                         const float* n_sum, const float* e_sum, float* g_out,
                         float invN, float inv2E)
{
  __shared__ float x[96];
  int t = threadIdx.x;  // 32 threads
  if (t < 32) { x[t] = n_sum[t] * invN; x[32 + t] = e_sum[t] * inv2E; x[64 + t] = gf[t]; }
  __syncthreads();
  if (t < 32) {
    float s = beff[t];
    for (int j = 0; j < 96; ++j) s += Weff[t * 96 + j] * x[j];
    g_out[t] = s + gf[t];
  }
}

// ----------------------------------------------------------------- set2set ----
__global__ void k_lstm(float* h, float* c, float* q, float* wsum, float* esum,
                       const float* Wih, const float* Whh,
                       const float* bih, const float* bhh, int step)
{
  __shared__ float sg[128], sh[32], sq[64];
  int t = threadIdx.x;  // 128 threads
  if (step > 0) {
    if (t < 32) q[t] = h[t];
    else if (t < 64) q[t] = wsum[t - 32] / esum[0];
  }
  __syncthreads();
  if (t < 64) sq[t] = q[t];
  if (t < 32) sh[t] = h[t];
  __syncthreads();
  float g = bih[t] + bhh[t];
  for (int j = 0; j < 64; ++j) g += Wih[t * 64 + j] * sq[j];
  for (int j = 0; j < 32; ++j) g += Whh[t * 32 + j] * sh[j];
  sg[t] = g;
  __syncthreads();
  if (t < 32) {
    float ig = 1.f / (1.f + __expf(-sg[t]));
    float fg = 1.f / (1.f + __expf(-sg[32 + t]));
    float gg = tanhf(sg[64 + t]);
    float og = 1.f / (1.f + __expf(-sg[96 + t]));
    float cc = fg * c[t] + ig * gg;
    c[t] = cc;
    h[t] = og * tanhf(cc);
    wsum[t] = 0.f;           // reset attention accumulators for this step
    if (t == 0) esum[0] = 0.f;
  }
}

// softmax(x@h) attention + weighted sum, one wave per row, partials flushed
// with global atomics.
__global__ void __launch_bounds__(256) k_att(const float* __restrict__ x, int n,
                                             const float* __restrict__ h,
                                             float* wsum, float* esum)
{
  const int lane = threadIdx.x & 31;
  const int w  = (int)((blockIdx.x * blockDim.x + threadIdx.x) >> 5);
  const int nw = (int)((gridDim.x * blockDim.x) >> 5);
  const float hv = h[lane];
  float accv = 0.f, aes = 0.f;
  for (int row = w; row < n; row += nw) {
    float xv = x[(size_t)row * 32 + lane];
    float d = xv * hv;
    for (int off = 16; off > 0; off >>= 1) d += __shfl_xor(d, off, 32);
    float ev = __expf(d);
    accv += ev * xv;
    aes  += ev;
  }
  atomicAdd(&wsum[lane], accv);
  if (lane == 0) atomicAdd(esum, aes);
}

// --------------------------------------------------------------- head ----
__global__ void k_final(const float* hN, const float* wsN, const float* esN,
                        const float* hE, const float* wsE, const float* esE,
                        const float* gout,
                        const float* W1, const float* b1,
                        const float* W2, const float* b2,
                        const float* W3, const float* b3, float* out)
{
  __shared__ float v[160], d1[32], d2[16];
  int t = threadIdx.x;
  if (t < 32) {
    v[t]       = hN[t];
    v[32 + t]  = wsN[t] / esN[0];
    v[64 + t]  = hE[t];
    v[96 + t]  = wsE[t] / esE[0];
    v[128 + t] = gout[t];
  }
  __syncthreads();
  if (t < 32) {
    float s = b1[t];
    for (int j = 0; j < 160; ++j) s += W1[t * 160 + j] * v[j];
    d1[t] = s;
  }
  __syncthreads();
  if (t < 16) {
    float s = b2[t];
    for (int j = 0; j < 32; ++j) s += W2[t * 32 + j] * d1[j];
    d2[t] = s;
  }
  __syncthreads();
  if (t == 0) {
    float s = b3[0];
    for (int j = 0; j < 16; ++j) s += W3[j] * d2[j];
    out[0] = s;
  }
}

// ------------------------------------------------------------------ launch ---
extern "C" void kernel_launch(void* const* d_in, const int* in_sizes, int n_in,
                              void* d_out, int out_size, void* d_ws, size_t ws_size,
                              hipStream_t stream)
{
  (void)n_in; (void)out_size; (void)ws_size;
  const float* nf = (const float*)d_in[0];
  const int*   ei = (const int*)d_in[1];       // (2,E): [src(E) | dst(E)]
  const float* ef = (const float*)d_in[2];
  const float* gf = (const float*)d_in[3];
  // params, JAX pytree order (dict keys sorted):
  // blocks[0]: edge W0,b0,W1,b1,W2,b2 | global ... | node ...
  const float* eW0 = (const float*)d_in[4];  const float* eb0 = (const float*)d_in[5];
  const float* eW1 = (const float*)d_in[6];  const float* eb1 = (const float*)d_in[7];
  const float* eW2 = (const float*)d_in[8];  const float* eb2 = (const float*)d_in[9];
  const float* gW0 = (const float*)d_in[10]; const float* gb0 = (const float*)d_in[11];
  const float* gW1 = (const float*)d_in[12]; const float* gb1 = (const float*)d_in[13];
  const float* gW2 = (const float*)d_in[14]; const float* gb2 = (const float*)d_in[15];
  const float* nW0 = (const float*)d_in[16]; const float* nb0 = (const float*)d_in[17];
  const float* nW1 = (const float*)d_in[18]; const float* nb1 = (const float*)d_in[19];
  const float* nW2 = (const float*)d_in[20]; const float* nb2 = (const float*)d_in[21];
  const float* d1W = (const float*)d_in[22]; const float* d1b = (const float*)d_in[23];
  const float* d2W = (const float*)d_in[24]; const float* d2b = (const float*)d_in[25];
  const float* outW = (const float*)d_in[26]; const float* outB = (const float*)d_in[27];
  // s2s_edges: Whh, Wih, bhh, bih ; then s2s_nodes same
  const float* seWhh = (const float*)d_in[28]; const float* seWih = (const float*)d_in[29];
  const float* sebhh = (const float*)d_in[30]; const float* sebih = (const float*)d_in[31];
  const float* snWhh = (const float*)d_in[32]; const float* snWih = (const float*)d_in[33];
  const float* snbhh = (const float*)d_in[34]; const float* snbih = (const float*)d_in[35];

  const int N = in_sizes[0] / 32;
  const int E = in_sizes[2] / 32;
  const int* srcI = ei;
  const int* dstI = ei + E;

  float* ws = (float*)d_ws;
  size_t oACC   = 0;
  size_t oCNT   = oACC + (size_t)N * 32;
  size_t oSMALL = oCNT + (size_t)N;
  size_t oWt    = oSMALL + 1024;
  size_t oNF    = oWt + 16384;
  size_t oEF    = oNF + (size_t)N * 32;

  float* acc = ws + oACC;
  float* cnt = ws + oCNT;
  float* sm  = ws + oSMALL;
  float* WE  = ws + oWt;  float* BE = WE + 4096;
  float* WN  = BE + 64;   float* BN = WN + 3072;
  float* WG  = BN + 64;   float* BG = WG + 3072;
  float* nfo = ws + oNF;
  float* efo = ws + oEF;

  float* e_sum = sm + 0;   float* n_sum = sm + 32;
  float* hN = sm + 64;  float* cN = sm + 96;  float* qN = sm + 128;
  float* wsN = sm + 192; float* esN = sm + 224;
  float* hE = sm + 256; float* cE = sm + 288; float* qE = sm + 320;
  float* wsE = sm + 384; float* esE = sm + 416;
  float* gOut = sm + 448;

  long zeroN = (long)N * 32 + N + 1024;   // acc + cnt + small state block
  k_zero<<<2048, 256, 0, stream>>>(ws, zeroN);

  MlpW mE{eW0, eb0, eW1, eb1, eW2, eb2, WE, BE, 128};
  MlpW mN{nW0, nb0, nW1, nb1, nW2, nb2, WN, BN, 96};
  MlpW mG{gW0, gb0, gW1, gb1, gW2, gb2, WG, BG, 96};
  k_setup<<<1, 256, 0, stream>>>(mE, mN, mG);

  int eBlocks = (E + 127) / 128;
  k_edge<<<eBlocks, 256, 0, stream>>>(nf, srcI, dstI, ef, gf, WE, BE,
                                      acc, cnt, efo, e_sum, E);
  int nBlocks = (N + 127) / 128;
  k_node<<<nBlocks, 256, 0, stream>>>(nf, acc, cnt, gf, WN, BN, nfo, n_sum, N);
  k_global<<<1, 32, 0, stream>>>(gf, WG, BG, n_sum, e_sum, gOut,
                                 1.f / (float)N, 0.5f / (float)E);

  for (int s = 0; s < 3; ++s) {
    k_lstm<<<1, 128, 0, stream>>>(hN, cN, qN, wsN, esN, snWih, snWhh, snbih, snbhh, s);
    k_att<<<512, 256, 0, stream>>>(nfo, N, hN, wsN, esN);
  }
  for (int s = 0; s < 3; ++s) {
    k_lstm<<<1, 128, 0, stream>>>(hE, cE, qE, wsE, esE, seWih, seWhh, sebih, sebhh, s);
    k_att<<<512, 256, 0, stream>>>(efo, E, hE, wsE, esE);
  }
  k_final<<<1, 256, 0, stream>>>(hN, wsN, esN, hE, wsE, esE, gOut,
                                 d1W, d1b, d2W, d2b, outW, outB, (float*)d_out);
}
